// LSTM_944892805185
// MI455X (gfx1250) — matrix operand order repaired and checked by reading, not compiled, not run on hardware
//
#include <hip/hip_runtime.h>
#include <hip/hip_bf16.h>

typedef __attribute__((ext_vector_type(16))) __bf16 v16bf;
typedef __attribute__((ext_vector_type(8)))  float  v8f;
typedef __attribute__((ext_vector_type(4)))  int    v4i;

typedef __attribute__((address_space(1))) v4i gv4i;   // global int4*
typedef __attribute__((address_space(3))) v4i lv4i;   // LDS int4*

#define EMB     256
#define HID     516
#define HP      528      // padded hidden, 33*16
#define NT_H    33       // N tiles per gate (528/16)
#define KT_H    17       // K tiles over hidden (544/32)
#define KT_E    8        // K tiles over embedding (256/32)
#define BATCH   32
#define STEPS   64
#define VY      32000
#define NT_V    2000     // 32000/16
#define MT_ALL  128      // 2048/16
#define NBLK_PROJ 250    // 2000 ntiles / 8 waves

#define V16(p) (*(const v16bf*)(p))

#if __has_builtin(__builtin_amdgcn_global_load_async_to_lds_b128)
#define HAS_ASYNC_LDS 1
#else
#define HAS_ASYNC_LDS 0
#endif

__device__ __forceinline__ void wait_async0() {
#if __has_builtin(__builtin_amdgcn_s_wait_asynccnt)
  __builtin_amdgcn_s_wait_asynccnt(0);
#else
  asm volatile("s_wait_asynccnt 0x0" ::: "memory");
#endif
}

__device__ __forceinline__ v8f wmma_bf16(v16bf a, v16bf b, v8f c) {
  return __builtin_amdgcn_wmma_f32_16x16x32_bf16(false, a, false, b, (short)0, c, false, false);
}
__device__ __forceinline__ float sigm(float x) { return 1.0f / (1.0f + __expf(-x)); }

// ---------------------------------------------------------------------------
// Pack an fp32 weight matrix [G*Nreal rows, Kreal cols] (row = gate*Nreal+n)
// into bf16 WMMA B-fragment tiles: dst[(((g*NT+nt)*KT)+kt)*512 + lane*16 + j*2 + h]
// B layout: lanes 0-15 col=lane, K = {0..7 (j<4), 16..23 (j>=4)} ; lanes 16-31 col=lane-16, +8
// ---------------------------------------------------------------------------
__global__ __launch_bounds__(256)
void pack_b(const float* __restrict__ src, __bf16* __restrict__ dst,
            int Kreal, int KT, int Nreal, int NT, int G, int total) {
  int idx = blockIdx.x * 256 + threadIdx.x;
  if (idx >= total) return;
  int t2   = idx & 511;
  int tile = idx >> 9;
  int kt   = tile % KT;
  int tmp  = tile / KT;
  int nt   = tmp % NT;
  int g    = tmp / NT;
  int lane = t2 >> 4;
  int rr   = t2 & 15;
  int j    = rr >> 1, hh = rr & 1;
  int kl   = ((j < 4) ? (j * 2 + hh) : (16 + (j - 4) * 2 + hh)) + ((lane < 16) ? 0 : 8);
  int k    = kt * 32 + kl;
  int n    = nt * 16 + (lane & 15);
  float v = 0.0f;
  if (k < Kreal && n < Nreal) v = src[(size_t)(g * Nreal + n) * Kreal + k];
  dst[idx] = (__bf16)v;
}

// ---------------------------------------------------------------------------
// Gather token embeddings and pack into bf16 WMMA A-fragment tiles.
// dst tile index = (t*2 + mt)*KT_E + kt ; within-tile = lane*16 + j*2 + h
// A layout: lanes 0-15 row=lane, K = {0..7 (j<4), 16..23 (j>=4)} ; lanes 16-31 +8
// ---------------------------------------------------------------------------
__global__ __launch_bounds__(256)
void embed_pack(const int* __restrict__ tok, const float* __restrict__ emb,
                __bf16* __restrict__ dst, int total) {
  int idx = blockIdx.x * 256 + threadIdx.x;
  if (idx >= total) return;
  int t2   = idx & 511;
  int tile = idx >> 9;
  int kt   = tile & 7;
  int tm   = tile >> 3;
  int mt   = tm & 1;
  int t    = tm >> 1;
  int lane = t2 >> 4;
  int rr   = t2 & 15;
  int j    = rr >> 1, hh = rr & 1;
  int kl = ((j < 4) ? (j * 2 + hh) : (16 + (j - 4) * 2 + hh)) + ((lane < 16) ? 0 : 8);
  int k  = kt * 32 + kl;
  int m  = mt * 16 + (lane & 15);
  int tk = tok[t * BATCH + m];
  dst[idx] = (__bf16)emb[(size_t)tk * EMB + k];
}

// b = bih + bhh, zero-padded to [4][528]
__global__ __launch_bounds__(256)
void pack_bias(const float* __restrict__ bih, const float* __restrict__ bhh,
               float* __restrict__ dst) {
  int idx = blockIdx.x * 256 + threadIdx.x;
  if (idx >= 4 * HP) return;
  int g = idx / HP, n = idx % HP;
  dst[idx] = (n < HID) ? (bih[g * HID + n] + bhh[g * HID + n]) : 0.0f;
}

// ---------------------------------------------------------------------------
// Persistent single-workgroup 2-layer LSTM recurrence (encoder then decoder).
// 32 waves; each wave owns fixed 16x16 hidden tiles per layer; c stays in VGPRs,
// h stays in LDS in WMMA A-fragment layout. Weight B-fragments are re-streamed
// from L2 every step through an opaque-offset pointer so the compiler cannot
// hoist them out of the time loop and spill them to scratch.
// ---------------------------------------------------------------------------
__global__ __launch_bounds__(1024)
void lstm_rec(const __bf16* __restrict__ XeF,  const __bf16* __restrict__ XdF,
              const __bf16* __restrict__ Wih0e, const __bf16* __restrict__ Whh0e,
              const __bf16* __restrict__ Wih1e, const __bf16* __restrict__ Whh1e,
              const __bf16* __restrict__ Wih0d, const __bf16* __restrict__ Whh0d,
              const __bf16* __restrict__ Wih1d, const __bf16* __restrict__ Whh1d,
              const float* __restrict__ b0e, const float* __restrict__ b1e,
              const float* __restrict__ b0d, const float* __restrict__ b1d,
              __bf16* __restrict__ H1F) {
  __shared__ __bf16 h0f[2 * KT_H * 512];   // 34816 B
  __shared__ __bf16 h1f[2 * KT_H * 512];   // 34816 B
  const int tid  = threadIdx.x;
  const int lane = tid & 31;
  const int wv   = tid >> 5;

  for (int i = tid; i < 2 * KT_H * 512; i += 1024) { h0f[i] = (__bf16)0.0f; h1f[i] = (__bf16)0.0f; }
  // zero the never-written pad stripe of H1F (kt==16, k'>=16 slots)
  for (int i = tid; i < MT_ALL * 512; i += 1024)
    H1F[(size_t)(i >> 9) * (KT_H * 512) + 16 * 512 + (i & 511)] = (__bf16)0.0f;
  __syncthreads();

  float c0[3][8], c1[3][8], hr[3][8];
#pragma unroll
  for (int s = 0; s < 3; ++s)
#pragma unroll
    for (int r = 0; r < 8; ++r) { c0[s][r] = 0.0f; c1[s][r] = 0.0f; }

  const int col = lane & 15;
  const int mhi = (lane >> 4) << 3;
  const v8f z8 = {0.f, 0.f, 0.f, 0.f, 0.f, 0.f, 0.f, 0.f};

  for (int phase = 0; phase < 2; ++phase) {
    const __bf16* XF  = phase ? XdF   : XeF;
    const __bf16* Wi0 = phase ? Wih0d : Wih0e;
    const __bf16* Wh0 = phase ? Whh0d : Whh0e;
    const __bf16* Wi1 = phase ? Wih1d : Wih1e;
    const __bf16* Wh1 = phase ? Whh1d : Whh1e;
    const float*  B0  = phase ? b0d : b0e;
    const float*  B1  = phase ? b1d : b1e;

    for (int t = 0; t < STEPS; ++t) {
      // Opaque zero: makes weight addresses loop-variant so the B-fragment
      // global loads stay inside the t-loop (L2 hits) instead of being
      // LICM-hoisted and spilled to scratch.
      int opq = 0;
      asm volatile("" : "+s"(opq));
      const __bf16* Wi0t = Wi0 + opq;
      const __bf16* Wh0t = Wh0 + opq;
      const __bf16* Wi1t = Wi1 + opq;
      const __bf16* Wh1t = Wh1 + opq;

      // -------- layer 0: gates = X[t] @ Wih0^T + h0 @ Whh0^T --------
#pragma unroll
      for (int s = 0; s < 3; ++s) {
        const int u = wv + 32 * s;
        if (u < 2 * NT_H) {
          const int nt = u >> 1, mt = u & 1;
          v8f ac[4] = {z8, z8, z8, z8};
          for (int kt = 0; kt < KT_E; ++kt) {
            v16bf av = V16(&XF[(size_t)((t * 2 + mt) * KT_E + kt) * 512 + lane * 16]);
#pragma unroll
            for (int g = 0; g < 4; ++g)
              ac[g] = wmma_bf16(av, V16(&Wi0t[(size_t)((g * NT_H + nt) * KT_E + kt) * 512 + lane * 16]), ac[g]);
          }
          for (int kt = 0; kt < KT_H; ++kt) {
            v16bf av = V16(&h0f[(mt * KT_H + kt) * 512 + lane * 16]);
#pragma unroll
            for (int g = 0; g < 4; ++g)
              ac[g] = wmma_bf16(av, V16(&Wh0t[(size_t)((g * NT_H + nt) * KT_H + kt) * 512 + lane * 16]), ac[g]);
          }
          const int n = nt * 16 + col;
          const float bI = B0[n], bF = B0[HP + n], bG = B0[2 * HP + n], bO = B0[3 * HP + n];
#pragma unroll
          for (int r = 0; r < 8; ++r) {
            float cv = sigm(ac[1][r] + bF) * c0[s][r] + sigm(ac[0][r] + bI) * tanhf(ac[2][r] + bG);
            c0[s][r] = cv;
            hr[s][r] = sigm(ac[3][r] + bO) * tanhf(cv);
          }
        }
      }
      __syncthreads();   // everyone done reading old h0f
      // -------- scatter new h0 into A-fragment layout --------
#pragma unroll
      for (int s = 0; s < 3; ++s) {
        const int u = wv + 32 * s;
        if (u < 2 * NT_H) {
          const int nt = u >> 1, mt = u & 1;
          const int n = nt * 16 + col;
          const int kt2 = n >> 5, kp = n & 31;
          const int jj = ((kp >> 4) << 2) + ((kp & 7) >> 1);
          const int hh = kp & 1;
          const int ls = ((kp >> 3) & 1) << 4;
#pragma unroll
          for (int r = 0; r < 8; ++r)
            h0f[(mt * KT_H + kt2) * 512 + (mhi + r + ls) * 16 + jj * 2 + hh] = (__bf16)hr[s][r];
        }
      }
      __syncthreads();   // new h0f visible
      // -------- layer 1: gates = h0 @ Wih1^T + h1 @ Whh1^T --------
#pragma unroll
      for (int s = 0; s < 3; ++s) {
        const int u = wv + 32 * s;
        if (u < 2 * NT_H) {
          const int nt = u >> 1, mt = u & 1;
          v8f ac[4] = {z8, z8, z8, z8};
          for (int kt = 0; kt < KT_H; ++kt) {
            v16bf av = V16(&h0f[(mt * KT_H + kt) * 512 + lane * 16]);
#pragma unroll
            for (int g = 0; g < 4; ++g)
              ac[g] = wmma_bf16(av, V16(&Wi1t[(size_t)((g * NT_H + nt) * KT_H + kt) * 512 + lane * 16]), ac[g]);
          }
          for (int kt = 0; kt < KT_H; ++kt) {
            v16bf av = V16(&h1f[(mt * KT_H + kt) * 512 + lane * 16]);
#pragma unroll
            for (int g = 0; g < 4; ++g)
              ac[g] = wmma_bf16(av, V16(&Wh1t[(size_t)((g * NT_H + nt) * KT_H + kt) * 512 + lane * 16]), ac[g]);
          }
          const int n = nt * 16 + col;
          const float bI = B1[n], bF = B1[HP + n], bG = B1[2 * HP + n], bO = B1[3 * HP + n];
#pragma unroll
          for (int r = 0; r < 8; ++r) {
            float cv = sigm(ac[1][r] + bF) * c1[s][r] + sigm(ac[0][r] + bI) * tanhf(ac[2][r] + bG);
            c1[s][r] = cv;
            hr[s][r] = sigm(ac[3][r] + bO) * tanhf(cv);
          }
        }
      }
      __syncthreads();   // everyone done reading old h1f
      // -------- scatter new h1 (and to global during decode) --------
#pragma unroll
      for (int s = 0; s < 3; ++s) {
        const int u = wv + 32 * s;
        if (u < 2 * NT_H) {
          const int nt = u >> 1, mt = u & 1;
          const int n = nt * 16 + col;
          const int kt2 = n >> 5, kp = n & 31;
          const int jj = ((kp >> 4) << 2) + ((kp & 7) >> 1);
          const int hh = kp & 1;
          const int ls = ((kp >> 3) & 1) << 4;
#pragma unroll
          for (int r = 0; r < 8; ++r) {
            const int eidx = (mhi + r + ls) * 16 + jj * 2 + hh;
            __bf16 hv = (__bf16)hr[s][r];
            h1f[(mt * KT_H + kt2) * 512 + eidx] = hv;
            if (phase) H1F[(size_t)((2 * t + mt) * KT_H + kt2) * 512 + eidx] = hv;
          }
        }
      }
      __syncthreads();
    }
  }
}

// ---------------------------------------------------------------------------
// Fused projection + streaming cross-entropy partials.
// 250 blocks x 8 waves. Each wave owns one 16-col vocab tile and keeps its 17
// Wout B-fragments resident in VGPRs (reused across all 128 M-tiles). The A
// tiles (shared by all 8 waves) are double-buffered in LDS: the next M-tile is
// staged with GLOBAL_LOAD_ASYNC_TO_LDS (ASYNCcnt) while WMMAs consume the
// current one, hiding all staging latency behind the matrix pipe.
// ---------------------------------------------------------------------------
__global__ __launch_bounds__(256)
void proj_ce(const __bf16* __restrict__ H1F, const __bf16* __restrict__ WoutP,
             const float* __restrict__ bout, const int* __restrict__ y,
             float* __restrict__ partial, float* __restrict__ tgt) {
  __shared__ __bf16 atile[2][KT_H * 512];   // 2 x 17 KB double buffer
  const int tid   = threadIdx.x;
  const int lane  = tid & 31;
  const int wv    = tid >> 5;
  const int ntile = blockIdx.x * 8 + wv;
  const int colg  = ntile * 16 + (lane & 15);
  const float bias = bout[colg];
  const int mhi = (lane >> 4) << 3;
  const v8f z8 = {0.f, 0.f, 0.f, 0.f, 0.f, 0.f, 0.f, 0.f};

  // Hold this wave's 17 B-fragments in registers (17 x 8 = 136 VGPRs).
  v16bf breg[KT_H];
#pragma unroll
  for (int kt = 0; kt < KT_H; ++kt)
    breg[kt] = V16(&WoutP[(size_t)(ntile * KT_H + kt) * 512 + lane * 16]);

  const uint4* asrc = (const uint4*)H1F;    // 16B chunks; 1088 chunks per M-tile

  // prologue: stage M-tile 0 into buffer 0
#if HAS_ASYNC_LDS
  for (int i = tid; i < 1088; i += 256)
    __builtin_amdgcn_global_load_async_to_lds_b128(
        (gv4i*)(asrc + i), (lv4i*)((uint4*)&atile[0][0] + i), 0, 0);
  wait_async0();
#else
  for (int i = tid; i < 1088; i += 256)
    *((uint4*)&atile[0][0] + i) = asrc[i];
#endif
  __syncthreads();

  for (int mt = 0; mt < MT_ALL; ++mt) {
    const __bf16* cur = &atile[mt & 1][0];
    // kick off async stage of the next M-tile into the other buffer
    if (mt + 1 < MT_ALL) {
      uint4* nxt = (uint4*)&atile[(mt + 1) & 1][0];
#if HAS_ASYNC_LDS
      for (int i = tid; i < 1088; i += 256)
        __builtin_amdgcn_global_load_async_to_lds_b128(
            (gv4i*)(asrc + (size_t)(mt + 1) * 1088 + i), (lv4i*)(nxt + i), 0, 0);
#else
      for (int i = tid; i < 1088; i += 256)
        nxt[i] = asrc[(size_t)(mt + 1) * 1088 + i];
#endif
    }

    v8f acc = z8;
#pragma unroll
    for (int kt = 0; kt < KT_H; ++kt)
      acc = wmma_bf16(V16(&cur[kt * 512 + lane * 16]), breg[kt], acc);

#pragma unroll
    for (int r = 0; r < 8; ++r) {
      float v = acc[r] + bias;
      int row = mt * 16 + mhi + r;              // row = step*32 + batch
      int tok = y[BATCH + row];                 // y[(step+1)*32 + batch]
      if (tok == colg) tgt[row] = v;            // unique writer per row
      float e = __expf(v);                      // logits are tame (|v| small)
      e += __shfl_xor(e, 1);
      e += __shfl_xor(e, 2);
      e += __shfl_xor(e, 4);
      e += __shfl_xor(e, 8);                    // sum over this tile's 16 cols
      if ((lane & 15) == 0) partial[(size_t)blockIdx.x * 2048 + row] = e;
    }
#if HAS_ASYNC_LDS
    if (mt + 1 < MT_ALL) wait_async0();         // my async stores to LDS landed
#endif
    __syncthreads();                            // everyone's staging visible
  }
}

// Deterministic final reduction: loss = (1/32) * sum_rows( log(sum exp) - logit_tgt )
__global__ __launch_bounds__(256)
void ce_reduce(const float* __restrict__ partial, const float* __restrict__ tgt,
               float* __restrict__ out) {
  __shared__ float red[256];
  float acc = 0.0f;
  for (int row = threadIdx.x; row < 2048; row += 256) {
    float s = 0.0f;
    for (int b = 0; b < NBLK_PROJ; ++b) s += partial[(size_t)b * 2048 + row];
    acc += __logf(s) - tgt[row];
  }
  red[threadIdx.x] = acc;
  __syncthreads();
  for (int off = 128; off > 0; off >>= 1) {
    if (threadIdx.x < off) red[threadIdx.x] += red[threadIdx.x + off];
    __syncthreads();
  }
  if (threadIdx.x == 0) out[0] = red[0] * (1.0f / 32.0f);
}

// ---------------------------------------------------------------------------
extern "C" void kernel_launch(void* const* d_in, const int* in_sizes, int n_in,
                              void* d_out, int out_size, void* d_ws, size_t ws_size,
                              hipStream_t stream) {
  const int*   x      = (const int*)d_in[0];
  const int*   y      = (const int*)d_in[1];
  const float* encemb = (const float*)d_in[2];
  const float* decemb = (const float*)d_in[3];
  const float* eWih0  = (const float*)d_in[4];
  const float* eWhh0  = (const float*)d_in[5];
  const float* eBih0  = (const float*)d_in[6];
  const float* eBhh0  = (const float*)d_in[7];
  const float* eWih1  = (const float*)d_in[8];
  const float* eWhh1  = (const float*)d_in[9];
  const float* eBih1  = (const float*)d_in[10];
  const float* eBhh1  = (const float*)d_in[11];
  const float* dWih0  = (const float*)d_in[12];
  const float* dWhh0  = (const float*)d_in[13];
  const float* dBih0  = (const float*)d_in[14];
  const float* dBhh0  = (const float*)d_in[15];
  const float* dWih1  = (const float*)d_in[16];
  const float* dWhh1  = (const float*)d_in[17];
  const float* dBih1  = (const float*)d_in[18];
  const float* dBhh1  = (const float*)d_in[19];
  const float* Wout   = (const float*)d_in[20];
  const float* bout   = (const float*)d_in[21];
  float* out = (float*)d_out;

  // ---- bump allocator over workspace (total ~55 MB) ----
  char* ws = (char*)d_ws;
  size_t off = 0;
  auto alloc = [&](size_t bytes) -> void* {
    void* p = ws + off;
    off = (off + bytes + 255) & ~(size_t)255;
    return p;
  };
  const size_t XF_ELEMS   = (size_t)STEPS * 2 * KT_E * 512;      // 524288
  const size_t WIH0_ELEMS = (size_t)4 * NT_H * KT_E * 512;       // 540672
  const size_t WHH_ELEMS  = (size_t)4 * NT_H * KT_H * 512;       // 1148928
  const size_t WOUT_ELEMS = (size_t)NT_V * KT_H * 512;           // 17408000
  const size_t H1F_ELEMS  = (size_t)MT_ALL * KT_H * 512;         // 1114112

  __bf16* XeF   = (__bf16*)alloc(XF_ELEMS * 2);
  __bf16* XdF   = (__bf16*)alloc(XF_ELEMS * 2);
  __bf16* Wih0e = (__bf16*)alloc(WIH0_ELEMS * 2);
  __bf16* Wih0d = (__bf16*)alloc(WIH0_ELEMS * 2);
  __bf16* Whh0e = (__bf16*)alloc(WHH_ELEMS * 2);
  __bf16* Wih1e = (__bf16*)alloc(WHH_ELEMS * 2);
  __bf16* Whh1e = (__bf16*)alloc(WHH_ELEMS * 2);
  __bf16* Whh0d = (__bf16*)alloc(WHH_ELEMS * 2);
  __bf16* Wih1d = (__bf16*)alloc(WHH_ELEMS * 2);
  __bf16* Whh1d = (__bf16*)alloc(WHH_ELEMS * 2);
  __bf16* WoutP = (__bf16*)alloc(WOUT_ELEMS * 2);
  float*  b0e   = (float*)alloc(4 * HP * 4);
  float*  b1e   = (float*)alloc(4 * HP * 4);
  float*  b0d   = (float*)alloc(4 * HP * 4);
  float*  b1d   = (float*)alloc(4 * HP * 4);
  __bf16* H1F   = (__bf16*)alloc(H1F_ELEMS * 2);
  float*  part  = (float*)alloc((size_t)NBLK_PROJ * 2048 * 4);
  float*  tgt   = (float*)alloc(2048 * 4);
  (void)ws_size; (void)in_sizes; (void)n_in; (void)out_size;

  // ---- weight / bias / embedding packing (parallel, order-free) ----
  auto nb = [](size_t total) { return (int)((total + 255) / 256); };
  pack_b<<<nb(WIH0_ELEMS), 256, 0, stream>>>(eWih0, Wih0e, EMB, KT_E, HID, NT_H, 4, (int)WIH0_ELEMS);
  pack_b<<<nb(WIH0_ELEMS), 256, 0, stream>>>(dWih0, Wih0d, EMB, KT_E, HID, NT_H, 4, (int)WIH0_ELEMS);
  pack_b<<<nb(WHH_ELEMS), 256, 0, stream>>>(eWhh0, Whh0e, HID, KT_H, HID, NT_H, 4, (int)WHH_ELEMS);
  pack_b<<<nb(WHH_ELEMS), 256, 0, stream>>>(eWih1, Wih1e, HID, KT_H, HID, NT_H, 4, (int)WHH_ELEMS);
  pack_b<<<nb(WHH_ELEMS), 256, 0, stream>>>(eWhh1, Whh1e, HID, KT_H, HID, NT_H, 4, (int)WHH_ELEMS);
  pack_b<<<nb(WHH_ELEMS), 256, 0, stream>>>(dWhh0, Whh0d, HID, KT_H, HID, NT_H, 4, (int)WHH_ELEMS);
  pack_b<<<nb(WHH_ELEMS), 256, 0, stream>>>(dWih1, Wih1d, HID, KT_H, HID, NT_H, 4, (int)WHH_ELEMS);
  pack_b<<<nb(WHH_ELEMS), 256, 0, stream>>>(dWhh1, Whh1d, HID, KT_H, HID, NT_H, 4, (int)WHH_ELEMS);
  pack_b<<<nb(WOUT_ELEMS), 256, 0, stream>>>(Wout, WoutP, HID, KT_H, VY, NT_V, 1, (int)WOUT_ELEMS);
  pack_bias<<<nb(4 * HP), 256, 0, stream>>>(eBih0, eBhh0, b0e);
  pack_bias<<<nb(4 * HP), 256, 0, stream>>>(eBih1, eBhh1, b1e);
  pack_bias<<<nb(4 * HP), 256, 0, stream>>>(dBih0, dBhh0, b0d);
  pack_bias<<<nb(4 * HP), 256, 0, stream>>>(dBih1, dBhh1, b1d);
  embed_pack<<<nb(XF_ELEMS), 256, 0, stream>>>(x, encemb, XeF, (int)XF_ELEMS);
  embed_pack<<<nb(XF_ELEMS), 256, 0, stream>>>(y, decemb, XdF, (int)XF_ELEMS);  // y[0..63]

  // ---- sequential recurrence: one persistent workgroup, 32 waves ----
  lstm_rec<<<1, 1024, 0, stream>>>(XeF, XdF,
                                   Wih0e, Whh0e, Wih1e, Whh1e,
                                   Wih0d, Whh0d, Wih1d, Whh1d,
                                   b0e, b1e, b0d, b1d, H1F);

  // ---- batched projection [2048 x 544] @ [544 x 32000] + fused CE ----
  proj_ce<<<NBLK_PROJ, 256, 0, stream>>>(H1F, WoutP, bout, y, part, tgt);
  ce_reduce<<<1, 256, 0, stream>>>(part, tgt, out);
}
